// GNNProcessor_69861938037048
// MI455X (gfx1250) — compile-verified, matrix operand order checked by reading, not run on hardware
//
#include <hip/hip_runtime.h>
#include <hip/hip_bf16.h>

// ---------------------------------------------------------------------------
// GNN transformer layer for MI455X (gfx1250, wave32).
// All dense GEMMs run through one bf16 WMMA kernel (v_wmma_f32_16x16x32_bf16),
// 4 M-tiles per wave for B-fragment reuse. Graph attention (scatter softmax +
// scatter add) uses fp32 global atomics.
// ---------------------------------------------------------------------------

typedef __bf16 bf16_t;
typedef __attribute__((ext_vector_type(16))) __bf16 v16bf;
typedef __attribute__((ext_vector_type(8)))  float  v8f;

// Convert 16 fp32 values (two 16B chunks per half-lane, per the 16-bit WMMA
// A-operand layout) to a bf16 A fragment with native cvt instructions.
__device__ __forceinline__ v16bf cvt_frag_bf16(float4 c0, float4 c1,
                                               float4 c2, float4 c3) {
  v16bf a;
  a[0]  = (__bf16)c0.x; a[1]  = (__bf16)c0.y; a[2]  = (__bf16)c0.z; a[3]  = (__bf16)c0.w;
  a[4]  = (__bf16)c1.x; a[5]  = (__bf16)c1.y; a[6]  = (__bf16)c1.z; a[7]  = (__bf16)c1.w;
  a[8]  = (__bf16)c2.x; a[9]  = (__bf16)c2.y; a[10] = (__bf16)c2.z; a[11] = (__bf16)c2.w;
  a[12] = (__bf16)c3.x; a[13] = (__bf16)c3.y; a[14] = (__bf16)c3.z; a[15] = (__bf16)c3.w;
  return a;
}

// ---------------------------------------------------------------------------
// Pack a row-major fp32 weight W[K][Ncols] into per-lane WMMA B fragments.
// Fragment layout: out[(((kt*nTiles + nt)*32 + lane)*16 + j)], where the
// (lane, j) -> (k, n) mapping matches the 16-bit WMMA operand layout:
//   n = nt*16 + (lane & 15)
//   k = kt*32 + (j & 7) + 8*((j >> 3)*2 + (lane >> 4))
// ---------------------------------------------------------------------------
__global__ void pack_b_kernel(const float* __restrict__ W,
                              bf16_t* __restrict__ out,
                              int K, int Ncols) {
  int idx = blockIdx.x * blockDim.x + threadIdx.x;
  int total = K * Ncols;
  if (idx >= total) return;
  int j    = idx & 15;
  int lane = (idx >> 4) & 31;
  int t    = idx >> 9;
  int nTiles = Ncols >> 4;
  int nt = t % nTiles;
  int kt = t / nTiles;
  int k  = kt * 32 + (j & 7) + 8 * (((j >> 3) << 1) + (lane >> 4));
  int n  = nt * 16 + (lane & 15);
  out[idx] = (__bf16)W[(size_t)k * Ncols + n];
}

// ---------------------------------------------------------------------------
// LayerNorm over D=128, one wave per row (wave32: 4 floats/lane).
// ---------------------------------------------------------------------------
__global__ void layernorm_kernel(const float* __restrict__ X,
                                 const float* __restrict__ g,
                                 const float* __restrict__ b,
                                 float* __restrict__ Y, int nrows) {
  int wave = (int)((blockIdx.x * (size_t)blockDim.x + threadIdx.x) >> 5);
  int lane = threadIdx.x & 31;
  if (wave >= nrows) return;
  const float4* row = reinterpret_cast<const float4*>(X + (size_t)wave * 128);
  float4 v = row[lane];
  float s = v.x + v.y + v.z + v.w;
#pragma unroll
  for (int off = 16; off; off >>= 1) s += __shfl_xor(s, off, 32);
  float mu = s * (1.0f / 128.0f);
  float dx = v.x - mu, dy = v.y - mu, dz = v.z - mu, dw = v.w - mu;
  float ss = dx * dx + dy * dy + dz * dz + dw * dw;
#pragma unroll
  for (int off = 16; off; off >>= 1) ss += __shfl_xor(ss, off, 32);
  float rstd = rsqrtf(ss * (1.0f / 128.0f) + 1e-5f);
  float4 gg = reinterpret_cast<const float4*>(g)[lane];
  float4 bb = reinterpret_cast<const float4*>(b)[lane];
  float4 o;
  o.x = dx * rstd * gg.x + bb.x;
  o.y = dy * rstd * gg.y + bb.y;
  o.z = dz * rstd * gg.z + bb.z;
  o.w = dw * rstd * gg.w + bb.w;
  reinterpret_cast<float4*>(Y + (size_t)wave * 128)[lane] = o;
}

// ---------------------------------------------------------------------------
// WMMA GEMM: C[M,Ncols] = act(A[M,K] * Bpacked + bias) + resid
// A fp32 row-major, converted to bf16 in-register. Each wave owns a 64x16
// output stripe (MT=4 row tiles) so each B fragment is reused 4x; block =
// 256 threads = 8 waves = 64 rows x 128 cols; gridDim.y covers Ncols/128.
// act: 0 = identity, 1 = SiLU (applied before residual add).
// ---------------------------------------------------------------------------
#define MT 4
__global__ void gemm_wmma_kernel(const float* __restrict__ A,
                                 const bf16_t* __restrict__ Bp,
                                 const float* __restrict__ bias,
                                 const float* __restrict__ resid,
                                 float* __restrict__ C,
                                 int M, int K, int Ncols, int act) {
  int wave = threadIdx.x >> 5;
  int lane = threadIdx.x & 31;
  int nTiles = Ncols >> 4;
  int nt = blockIdx.y * 8 + wave;
  if (nt >= nTiles) return;                    // wave-uniform: EXEC stays full
  int col  = nt * 16 + (lane & 15);
  int half = lane >> 4;
  int mBase = blockIdx.x * (16 * MT);

  const float* arow[MT];
#pragma unroll
  for (int i = 0; i < MT; ++i) {
    int r = mBase + i * 16 + (lane & 15);
    if (r >= M) r = M - 1;                     // clamp loads; stores guarded
    arow[i] = A + (size_t)r * K;
  }

  v8f acc[MT];
  float bv = bias ? bias[col] : 0.0f;
#pragma unroll
  for (int i = 0; i < MT; ++i)
#pragma unroll
    for (int r = 0; r < 8; ++r) acc[i][r] = bv;

  int kTiles = K >> 5;
#pragma unroll 4
  for (int kt = 0; kt < kTiles; ++kt) {
    int kb = kt * 32 + half * 8;
    const v16bf bfrag = *reinterpret_cast<const v16bf*>(
        Bp + (((size_t)kt * nTiles + nt) * 32 + lane) * 16);
#pragma unroll
    for (int i = 0; i < MT; ++i) {
      float4 c0 = *reinterpret_cast<const float4*>(arow[i] + kb + 0);
      float4 c1 = *reinterpret_cast<const float4*>(arow[i] + kb + 4);
      float4 c2 = *reinterpret_cast<const float4*>(arow[i] + kb + 16);
      float4 c3 = *reinterpret_cast<const float4*>(arow[i] + kb + 20);
      v16bf afrag = cvt_frag_bf16(c0, c1, c2, c3);
      acc[i] = __builtin_amdgcn_wmma_f32_16x16x32_bf16(
          /*neg_a=*/false, afrag, /*neg_b=*/false, bfrag,
          /*c_mod=*/(short)0, acc[i], /*reuse_a=*/false, /*reuse_b=*/false);
    }
  }

#pragma unroll
  for (int i = 0; i < MT; ++i) {
#pragma unroll
    for (int r = 0; r < 8; ++r) {
      int mrow = mBase + i * 16 + r + 8 * half;
      if (mrow < M) {
        float val = acc[i][r];
        if (act == 1) {                                       // SiLU, fast rcp
          float sig = __builtin_amdgcn_rcpf(1.0f + __expf(-val));
          val *= sig;
        }
        if (resid) val += resid[(size_t)mrow * Ncols + col];
        C[(size_t)mrow * Ncols + col] = val;
      }
    }
  }
}

// ---------------------------------------------------------------------------
// Scratch init: amax <- INT_MIN (encoded -inf floor), denom <- 0, agg <- 0.
// ---------------------------------------------------------------------------
__global__ void init_kernel(int* __restrict__ amax, float* __restrict__ denom,
                            float* __restrict__ agg, int nNodes) {
  int t = blockIdx.x * blockDim.x + threadIdx.x;
  int nh = nNodes * 8;
  if (t < nh) { amax[t] = (int)0x80000000; denom[t] = 0.0f; }
  int nd = nNodes * 128;
  if (t < nd) agg[t] = 0.0f;
}

__device__ __forceinline__ int enc_f32(float f) {
  int m = __float_as_int(f);
  return m >= 0 ? m : (m ^ 0x7fffffff);
}

// alpha logits per (edge, head) + per-dst atomic max (order-preserving int enc)
__global__ void alpha_kernel(const float* __restrict__ q,
                             const float* __restrict__ k,
                             const float* __restrict__ ep,
                             const int* __restrict__ src,
                             const int* __restrict__ dst,
                             float* __restrict__ alpha,
                             int* __restrict__ amax, int nE) {
  int t = blockIdx.x * blockDim.x + threadIdx.x;
  if (t >= nE * 8) return;
  int e = t >> 3, h = t & 7;
  int s = src[e], d = dst[e];
  const float4* qv = reinterpret_cast<const float4*>(q + (size_t)d * 128 + h * 16);
  const float4* kv = reinterpret_cast<const float4*>(k + (size_t)s * 128 + h * 16);
  const float4* ev = reinterpret_cast<const float4*>(ep + (size_t)e * 128 + h * 16);
  float acc = 0.0f;
#pragma unroll
  for (int i = 0; i < 4; ++i) {
    float4 a = qv[i], b = kv[i], c = ev[i];
    acc += a.x * (b.x + c.x) + a.y * (b.y + c.y) +
           a.z * (b.z + c.z) + a.w * (b.w + c.w);
  }
  acc *= 0.25f;                       // 1/sqrt(C), C=16
  alpha[t] = acc;
  atomicMax(&amax[d * 8 + h], enc_f32(acc));
}

// exp(alpha - amax[dst]) (neutralize non-finite max), accumulate denom
__global__ void exp_kernel(float* __restrict__ alpha,
                           const int* __restrict__ dst,
                           const int* __restrict__ amax,
                           float* __restrict__ denom, int nE) {
  int t = blockIdx.x * blockDim.x + threadIdx.x;
  if (t >= nE * 8) return;
  int e = t >> 3, h = t & 7;
  int d = dst[e];
  int enc = amax[d * 8 + h];
  int bits = enc >= 0 ? enc : (enc ^ 0x7fffffff);
  float m = __int_as_float(bits);
  if (((bits >> 23) & 0xFF) == 0xFF) m = 0.0f;   // inf/nan/empty -> 0
  float ex = __expf(alpha[t] - m);
  alpha[t] = ex;
  atomicAdd(&denom[d * 8 + h], ex);
}

// msg = (v[src] + e) * alpha/denom, scatter-add into agg[dst]
__global__ void agg_kernel(const float* __restrict__ vproj,
                           const float* __restrict__ ep,
                           const float* __restrict__ alpha,
                           const float* __restrict__ denom,
                           const int* __restrict__ src,
                           const int* __restrict__ dst,
                           float* __restrict__ agg, int nE) {
  int t = blockIdx.x * blockDim.x + threadIdx.x;
  if (t >= nE * 8) return;
  int e = t >> 3, h = t & 7;
  int s = src[e], d = dst[e];
  float w = alpha[t] * __builtin_amdgcn_rcpf(denom[d * 8 + h] + 1e-16f);
  const float4* vv = reinterpret_cast<const float4*>(vproj + (size_t)s * 128 + h * 16);
  const float4* ev = reinterpret_cast<const float4*>(ep + (size_t)e * 128 + h * 16);
  float* ag = agg + (size_t)d * 128 + h * 16;
#pragma unroll
  for (int i = 0; i < 4; ++i) {
    float4 a = vv[i], b = ev[i];
    atomicAdd(&ag[i * 4 + 0], (a.x + b.x) * w);
    atomicAdd(&ag[i * 4 + 1], (a.y + b.y) * w);
    atomicAdd(&ag[i * 4 + 2], (a.z + b.z) * w);
    atomicAdd(&ag[i * 4 + 3], (a.w + b.w) * w);
  }
}

// conv_out = agg + skip (elementwise, float4)
__global__ void add_kernel(const float* __restrict__ a, const float* __restrict__ b,
                           float* __restrict__ out, int n4) {
  int t = blockIdx.x * blockDim.x + threadIdx.x;
  if (t >= n4) return;
  float4 x = reinterpret_cast<const float4*>(a)[t];
  float4 y = reinterpret_cast<const float4*>(b)[t];
  float4 o; o.x = x.x + y.x; o.y = x.y + y.y; o.z = x.z + y.z; o.w = x.w + y.w;
  reinterpret_cast<float4*>(out)[t] = o;
}

// ---------------------------------------------------------------------------
// Host-side orchestration.
// Input order (setup_inputs dict, params flattened in dict order):
//  0:x 1:edge_index 2:edge_attr 3:Wq 4:bq 5:Wk 6:bk 7:Wv 8:bv 9:We
// 10:Wskip 11:bskip 12:Wproj 13:bproj 14:ln1_g 15:ln1_b 16:ln2_g 17:ln2_b
// 18:W1 19:b1 20:W2 21:b2
// ---------------------------------------------------------------------------
extern "C" void kernel_launch(void* const* d_in, const int* in_sizes, int n_in,
                              void* d_out, int out_size, void* d_ws, size_t ws_size,
                              hipStream_t stream) {
  const int D = 128, HID = 512;
  const int N = in_sizes[0] / D;        // 50000
  const int E = in_sizes[1] / 2;        // 800000

  const float* x      = (const float*)d_in[0];
  const int*   eidx   = (const int*)d_in[1];
  const int*   src    = eidx;
  const int*   dst    = eidx + E;
  const float* eattr  = (const float*)d_in[2];
  const float* Wq     = (const float*)d_in[3];  const float* bq    = (const float*)d_in[4];
  const float* Wk     = (const float*)d_in[5];  const float* bk    = (const float*)d_in[6];
  const float* Wv     = (const float*)d_in[7];  const float* bv    = (const float*)d_in[8];
  const float* We     = (const float*)d_in[9];
  const float* Wskip  = (const float*)d_in[10]; const float* bskip = (const float*)d_in[11];
  const float* Wproj  = (const float*)d_in[12]; const float* bproj = (const float*)d_in[13];
  const float* ln1g   = (const float*)d_in[14]; const float* ln1b  = (const float*)d_in[15];
  const float* ln2g   = (const float*)d_in[16]; const float* ln2b  = (const float*)d_in[17];
  const float* W1     = (const float*)d_in[18]; const float* b1    = (const float*)d_in[19];
  const float* W2     = (const float*)d_in[20]; const float* b2    = (const float*)d_in[21];

  // ---- workspace carve-up (fp32 unless noted) ----
  float* ws = (float*)d_ws;
  size_t NB = (size_t)N * D;
  float* xn    = ws;                         // [N,128]  (reused: conv_out)
  float* qb    = xn + NB;                    // [N,128]  (reused: out)
  float* kb    = qb + NB;                    // [N,128]  (reused: hn)
  float* vb    = kb + NB;                    // [N,128]
  float* skip  = vb + NB;                    // [N,128]
  float* h1    = skip + NB;                  // [N,512]
  float* ebuf  = h1 + (size_t)N * HID;       // [E,128]
  float* alpha = ebuf + (size_t)E * D;       // [E,8]
  int*   amax  = (int*)(alpha + (size_t)E * 8);   // [N,8]
  float* denom = (float*)(amax + (size_t)N * 8);  // [N,8]
  float* agg   = denom + (size_t)N * 8;      // [N,128]
  bf16_t* pw = (bf16_t*)(agg + NB);
  bf16_t* pWq    = pw;                       // 128*128 each
  bf16_t* pWk    = pWq    + 16384;
  bf16_t* pWv    = pWk    + 16384;
  bf16_t* pWskip = pWv    + 16384;
  bf16_t* pWe    = pWskip + 16384;
  bf16_t* pWproj = pWe    + 16384;
  bf16_t* pW1    = pWproj + 16384;           // 128*512
  bf16_t* pW2    = pW1    + 65536;           // 512*128

  float* conv = xn;    // alias after xn is dead
  float* out  = qb;    // alias after q is dead
  float* hn   = kb;    // alias after k is dead

  const int B256 = 256;
  dim3 blk(B256);

  // 1) pack weights into WMMA B-fragment layout (bf16)
  pack_b_kernel<<<dim3((128 * 128 + B256 - 1) / B256), blk, 0, stream>>>(Wq,    pWq,    128, 128);
  pack_b_kernel<<<dim3((128 * 128 + B256 - 1) / B256), blk, 0, stream>>>(Wk,    pWk,    128, 128);
  pack_b_kernel<<<dim3((128 * 128 + B256 - 1) / B256), blk, 0, stream>>>(Wv,    pWv,    128, 128);
  pack_b_kernel<<<dim3((128 * 128 + B256 - 1) / B256), blk, 0, stream>>>(Wskip, pWskip, 128, 128);
  pack_b_kernel<<<dim3((128 * 128 + B256 - 1) / B256), blk, 0, stream>>>(We,    pWe,    128, 128);
  pack_b_kernel<<<dim3((128 * 128 + B256 - 1) / B256), blk, 0, stream>>>(Wproj, pWproj, 128, 128);
  pack_b_kernel<<<dim3((128 * 512 + B256 - 1) / B256), blk, 0, stream>>>(W1,    pW1,    128, 512);
  pack_b_kernel<<<dim3((512 * 128 + B256 - 1) / B256), blk, 0, stream>>>(W2,    pW2,    512, 128);

  // 2) LayerNorm1: x -> xn  (8 rows per 256-thread block, wave per row)
  layernorm_kernel<<<dim3((N + 7) / 8), blk, 0, stream>>>(x, ln1g, ln1b, xn, N);

  // 3) node projections: q,k,v,skip = xn @ W + b   (M=N, K=128, Ncols=128)
  dim3 gN((N + 16 * MT - 1) / (16 * MT), 1);
  gemm_wmma_kernel<<<gN, blk, 0, stream>>>(xn, pWq,    bq,    nullptr, qb,   N, 128, 128, 0);
  gemm_wmma_kernel<<<gN, blk, 0, stream>>>(xn, pWk,    bk,    nullptr, kb,   N, 128, 128, 0);
  gemm_wmma_kernel<<<gN, blk, 0, stream>>>(xn, pWv,    bv,    nullptr, vb,   N, 128, 128, 0);
  gemm_wmma_kernel<<<gN, blk, 0, stream>>>(xn, pWskip, bskip, nullptr, skip, N, 128, 128, 0);

  // 4) edge projection: e = edge_attr @ We  (M=E) — the dominant GEMM
  dim3 gE((E + 16 * MT - 1) / (16 * MT), 1);
  gemm_wmma_kernel<<<gE, blk, 0, stream>>>(eattr, pWe, nullptr, nullptr, ebuf, E, 128, 128, 0);

  // 5) init scatter state (must rerun every call: no state across replays)
  init_kernel<<<dim3((N * 128 + B256 - 1) / B256), blk, 0, stream>>>(amax, denom, agg, N);

  // 6) logits + per-dst max
  int EH = E * 8;
  alpha_kernel<<<dim3((EH + B256 - 1) / B256), blk, 0, stream>>>(qb, kb, ebuf, src, dst, alpha, amax, E);

  // 7) exp + per-dst denom
  exp_kernel<<<dim3((EH + B256 - 1) / B256), blk, 0, stream>>>(alpha, dst, amax, denom, E);

  // 8) weighted scatter-add of messages
  agg_kernel<<<dim3((EH + B256 - 1) / B256), blk, 0, stream>>>(vb, ebuf, alpha, denom, src, dst, agg, E);

  // 9) conv_out = agg + skip
  add_kernel<<<dim3((N * 32 + B256 - 1) / B256), blk, 0, stream>>>(agg, skip, conv, N * 32);

  // 10) out = conv_out @ Wproj + bproj + x
  gemm_wmma_kernel<<<gN, blk, 0, stream>>>(conv, pWproj, bproj, x, out, N, 128, 128, 0);

  // 11) LayerNorm2: out -> hn
  layernorm_kernel<<<dim3((N + 7) / 8), blk, 0, stream>>>(out, ln2g, ln2b, hn, N);

  // 12) h1 = SiLU(hn @ W1 + b1)   (Ncols=512 -> grid.y=4)
  dim3 gM((N + 16 * MT - 1) / (16 * MT), 4);
  gemm_wmma_kernel<<<gM, blk, 0, stream>>>(hn, pW1, b1, nullptr, h1, N, 128, 512, 1);

  // 13) d_out = h1 @ W2 + b2 + out   (K=512)
  gemm_wmma_kernel<<<gN, blk, 0, stream>>>(h1, pW2, b2, out, (float*)d_out, N, 512, 128, 0);
}
